// CNN_LSTM_SE_7121055777258
// MI455X (gfx1250) — matrix operand order repaired and checked by reading, not compiled, not run on hardware
//
#include <hip/hip_runtime.h>
#include <hip/hip_bf16.h>

typedef __attribute__((ext_vector_type(16))) _Float16 v16h;
typedef __attribute__((ext_vector_type(8)))  _Float16 v8h;
typedef __attribute__((ext_vector_type(4)))  _Float16 v4h;
typedef __attribute__((ext_vector_type(8)))  float    v8f;

union F16Frag { v16h v; v8h h[2]; };

// A-matrix 16x32 f16 fragment (ISA 7.12.2): lanes 0-15 hold M=lane, K: VGPR0-3 -> K0-7,
// VGPR4-7 -> K16-23; lanes 16-31 shifted by +8 in K within each half.
__device__ __forceinline__ v16h load_frag_a(const _Float16* rowk, int hi) {
    F16Frag f;
    f.h[0] = *(const v8h*)(rowk + hi * 8);
    f.h[1] = *(const v8h*)(rowk + 16 + hi * 8);
    return f.v;
}
// B-matrix 32x16 f16 fragment: lane = N column; lanes 0-15 hold K0-7 and K16-23,
// lanes 16-31 shifted by +8 in K within each half (same K order as A).
__device__ __forceinline__ v16h load_frag_b(const _Float16* rowk, int hi) {
    F16Frag f;
    f.h[0] = *(const v8h*)(rowk + hi * 8);
    f.h[1] = *(const v8h*)(rowk + 16 + hi * 8);
    return f.v;
}

__device__ __forceinline__ float fast_sigmoid(float x) {
    return 1.0f / (1.0f + __expf(-x));
}
__device__ __forceinline__ float fast_tanh(float x) {
    return 1.0f - 2.0f / (__expf(2.0f * x) + 1.0f);
}

// ---------------------------------------------------------------- K0: convert
__global__ __launch_bounds__(256) void f32_to_f16_kernel(
    const float* __restrict__ src, _Float16* __restrict__ dst, int n4) {
    int i = blockIdx.x * blockDim.x + threadIdx.x;
    if (i < n4) {
        float4 v = ((const float4*)src)[i];
        v4h h;
        h[0] = (_Float16)v.x; h[1] = (_Float16)v.y;
        h[2] = (_Float16)v.z; h[3] = (_Float16)v.w;
        ((v4h*)dst)[i] = h;
    }
}

// ----------------------------------------------- K1: conv1x1+sigmoid+SE+pool
// One workgroup (8 wave32) per batch element. x[b] staged once into LDS f16;
// full 64x512 activation tile kept in LDS f32 for the SE/softmax/pool epilogue.
// Each wave owns a 16x32 output strip per pass: two WMMA accumulator chains
// share one A fragment (interleaved chains hide XDL latency, halve A traffic).
#define WW   64
#define DD   512
#define UU   512
#define XS_LD 520   // padded f16 row stride (1040 B -> rotates LDS banks)
#define CS_LD 516   // padded f32 row stride

__global__ __launch_bounds__(256) void conv_se_pool_kernel(
    const float* __restrict__ x,          // [B, 64, 512]
    const _Float16* __restrict__ convw,   // [512, 512] f16
    const float* __restrict__ convb,      // [512]
    const float* __restrict__ sew,        // [64, 64]
    const float* __restrict__ seb,        // [64]
    _Float16* __restrict__ pooled_h)      // [B, 512] f16
{
    extern __shared__ char smem[];
    _Float16* xs    = (_Float16*)smem;                  // 64 x 520 f16 = 66560 B
    float*    cs    = (float*)(smem + 66560);           // 64 x 516 f32 = 132096 B
    float*    avgS  = (float*)(smem + 198656);          // 64
    float*    lS    = avgS + 64;                        // 64
    float*    seS   = lS + 64;                          // 64
    float*    partS = seS + 64;                         // 256

    const int b    = blockIdx.x;
    const int tid  = threadIdx.x;
    const int wave = tid >> 5, lane = tid & 31;
    const int hi   = lane >> 4, ln = lane & 15;

    // Phase A: stage x[b] -> LDS as f16 (x read from HBM exactly once)
    const float4* xsrc = (const float4*)(x + (size_t)b * WW * DD);
    for (int i = tid; i < (WW * DD) / 4; i += 256) {
        float4 v = xsrc[i];
        int e = i * 4;
        int w = e >> 9, d = e & 511;
        v4h h;
        h[0] = (_Float16)v.x; h[1] = (_Float16)v.y;
        h[2] = (_Float16)v.z; h[3] = (_Float16)v.w;
        *(v4h*)(xs + w * XS_LD + d) = h;
    }
    __syncthreads();

    // Phase B: 64x512x512 GEMM via WMMA. 4x16 grid of 16x32 strips, 8 strips/wave.
    for (int t = 0; t < 8; ++t) {
        int T  = wave + (t << 3);          // 0..63
        int mt = T & 3, np = T >> 2;       // 4 M-tiles x 16 N-pairs
        int m0 = mt << 4, n0 = np << 5;
        v8f acc0 = {}, acc1 = {};
        const _Float16* arow  = xs + (m0 + ln) * XS_LD;
        const _Float16* brow0 = convw + (size_t)(n0 + ln) * DD;
        const _Float16* brow1 = convw + (size_t)(n0 + 16 + ln) * DD;
        for (int k0 = 0; k0 < DD; k0 += 32) {
            v16h a  = load_frag_a(arow + k0, hi);
            v16h b0 = load_frag_b(brow0 + k0, hi);
            v16h b1 = load_frag_b(brow1 + k0, hi);
            acc0 = __builtin_amdgcn_wmma_f32_16x16x32_f16(
                false, a, false, b0, (short)0, acc0, false, false);
            acc1 = __builtin_amdgcn_wmma_f32_16x16x32_f16(
                false, a, false, b1, (short)0, acc1, false, false);
        }
        float bias0 = convb[n0 + ln];
        float bias1 = convb[n0 + 16 + ln];
#pragma unroll
        for (int r = 0; r < 8; ++r) {
            int row = (m0 + r + hi * 8) * CS_LD;
            cs[row + n0 + ln]      = fast_sigmoid(acc0[r] + bias0);
            cs[row + n0 + 16 + ln] = fast_sigmoid(acc1[r] + bias1);
        }
    }
    __syncthreads();

    // Phase C: channel mean over U (all 256 threads produce quarter-row partials)
    {
        int w = tid >> 2, q = tid & 3;
        const float* row = cs + w * CS_LD + q * 128;
        float s = 0.f;
        for (int u = 0; u < 128; ++u) s += row[u];
        partS[tid] = s;
    }
    __syncthreads();
    if (tid < 64) {
        const float* p = partS + tid * 4;
        avgS[tid] = (p[0] + p[1] + p[2] + p[3]) * (1.0f / 512.0f);
    }
    __syncthreads();

    // Phase D: SE logits + softmax over the 64 window positions
    if (tid < 64) {
        float l = seb[tid];
        const float* wrow = sew + tid * 64;
        for (int j = 0; j < 64; ++j) l += avgS[j] * wrow[j];
        lS[tid] = l;
    }
    __syncthreads();
    if (tid < 64) {
        float m = -1e30f;
        for (int j = 0; j < 64; ++j) m = fmaxf(m, lS[j]);
        float den = 0.f;
        for (int j = 0; j < 64; ++j) den += __expf(lS[j] - m);
        seS[tid] = __expf(lS[tid] - m) / den;
    }
    __syncthreads();

    // Phase E: SE-weighted max-pool over window -> pooled[b, u] (f16 for next GEMM)
    for (int u = tid; u < UU; u += 256) {
        float mx = -1e30f;
        for (int w = 0; w < WW; ++w) mx = fmaxf(mx, cs[w * CS_LD + u] * seS[w]);
        pooled_h[(size_t)b * UU + u] = (_Float16)mx;
    }
}

// ------------------------------------------------ K2/K3: LSTM single step
// h0=c0=0 => gates = A @ w_ih^T + b_ih + b_hh, forget gate never used (c0=0).
// Each wave computes one (mt, nt, dir) triple: i/g/o gate 16x16 tiles share the
// A fragment; three interleaved WMMA chains. H=512, gate row offsets 0/1024/1536.
__global__ __launch_bounds__(256) void lstm_step_kernel(
    const _Float16* __restrict__ A,      // [2048, K] f16
    const _Float16* __restrict__ Wf,     // [2048, K] f16 (forward dir w_ih)
    const _Float16* __restrict__ Wr,     // [2048, K] f16 (reverse dir w_ih)
    const float* __restrict__ bihf, const float* __restrict__ bhhf,
    const float* __restrict__ bihr, const float* __restrict__ bhhr,
    int K,
    _Float16* __restrict__ out_h,        // [2048, 1024] f16 (or null)
    float* __restrict__ out_f,           // [2048, 1024] f32 (or null)
    int final_tanh)
{
    const int tid  = threadIdx.x;
    const int wave = tid >> 5, lane = tid & 31;
    const int hi   = lane >> 4, ln = lane & 15;

    int gw  = blockIdx.x * 8 + wave;     // 0..8191
    int dir = gw & 1;
    int nt  = (gw >> 1) & 31;
    int mt  = gw >> 6;                   // 0..127
    int m0  = mt << 4, n0 = nt << 4;

    const _Float16* Wm  = dir ? Wr : Wf;
    const float*    bih = dir ? bihr : bihf;
    const float*    bhh = dir ? bhhr : bhhf;

    v8f acc_i = {}, acc_g = {}, acc_o = {};
    const _Float16* arow  = A  + (size_t)(m0 + ln) * K;
    const _Float16* birow = Wm + (size_t)(n0 + ln) * K;           // i: rows [0,512)
    const _Float16* bgrow = Wm + (size_t)(1024 + n0 + ln) * K;    // g: rows [1024,1536)
    const _Float16* borow = Wm + (size_t)(1536 + n0 + ln) * K;    // o: rows [1536,2048)

    for (int k0 = 0; k0 < K; k0 += 32) {
        v16h a  = load_frag_a(arow + k0, hi);
        v16h bi = load_frag_b(birow + k0, hi);
        v16h bg = load_frag_b(bgrow + k0, hi);
        v16h bo = load_frag_b(borow + k0, hi);
        acc_i = __builtin_amdgcn_wmma_f32_16x16x32_f16(false, a, false, bi, (short)0, acc_i, false, false);
        acc_g = __builtin_amdgcn_wmma_f32_16x16x32_f16(false, a, false, bg, (short)0, acc_g, false, false);
        acc_o = __builtin_amdgcn_wmma_f32_16x16x32_f16(false, a, false, bo, (short)0, acc_o, false, false);
    }

    int n = n0 + ln;
    float bi_s = bih[n]        + bhh[n];
    float bg_s = bih[1024 + n] + bhh[1024 + n];
    float bo_s = bih[1536 + n] + bhh[1536 + n];
#pragma unroll
    for (int r = 0; r < 8; ++r) {
        float iv = fast_sigmoid(acc_i[r] + bi_s);
        float gv = fast_tanh(acc_g[r] + bg_s);
        float ov = fast_sigmoid(acc_o[r] + bo_s);
        float h  = ov * fast_tanh(iv * gv);
        if (final_tanh) h = fast_tanh(h);
        size_t idx = (size_t)(m0 + r + hi * 8) * 1024 + (size_t)dir * 512 + n;
        if (out_h) out_h[idx] = (_Float16)h;
        if (out_f) out_f[idx] = h;
    }
}

// ------------------------------------------------------------- K4: classifier
__global__ __launch_bounds__(256) void cls_kernel(
    const float* __restrict__ o1,    // [2048, 1024]
    const float* __restrict__ clsw,  // [1024]
    const float* __restrict__ clsb,  // [1]
    float* __restrict__ out)         // [2048]
{
    int wave = threadIdx.x >> 5, lane = threadIdx.x & 31;
    int b = blockIdx.x * 8 + wave;
    const float* row = o1 + (size_t)b * 1024;
    float s = 0.f;
    for (int j = lane; j < 1024; j += 32) s += row[j] * clsw[j];
#pragma unroll
    for (int off = 16; off > 0; off >>= 1) s += __shfl_xor(s, off, 32);
    if (lane == 0) out[b] = fast_tanh(s + clsb[0]);
}

extern "C" void kernel_launch(void* const* d_in, const int* in_sizes, int n_in,
                              void* d_out, int out_size, void* d_ws, size_t ws_size,
                              hipStream_t stream) {
    (void)in_sizes; (void)n_in; (void)out_size; (void)ws_size;

    const float* x        = (const float*)d_in[0];
    const float* conv_w   = (const float*)d_in[1];
    const float* conv_b   = (const float*)d_in[2];
    const float* se_w     = (const float*)d_in[3];
    const float* se_b     = (const float*)d_in[4];
    const float* w_ih_l0f = (const float*)d_in[5];
    const float* b_ih_l0f = (const float*)d_in[7];
    const float* b_hh_l0f = (const float*)d_in[8];
    const float* w_ih_l0r = (const float*)d_in[9];
    const float* b_ih_l0r = (const float*)d_in[11];
    const float* b_hh_l0r = (const float*)d_in[12];
    const float* w_ih_l1f = (const float*)d_in[13];
    const float* b_ih_l1f = (const float*)d_in[15];
    const float* b_hh_l1f = (const float*)d_in[16];
    const float* w_ih_l1r = (const float*)d_in[17];
    const float* b_ih_l1r = (const float*)d_in[19];
    const float* b_hh_l1r = (const float*)d_in[20];
    const float* cls_w    = (const float*)d_in[21];
    const float* cls_b    = (const float*)d_in[22];
    float* out = (float*)d_out;

    // Workspace layout (bytes). Total ~26.5 MB.
    char* ws = (char*)d_ws;
    _Float16* convw_h  = (_Float16*)(ws + 0);          // 512x512   f16 (0.5 MB)
    _Float16* w0f_h    = (_Float16*)(ws + 524288);     // 2048x512  f16 (2 MB)
    _Float16* w0r_h    = (_Float16*)(ws + 2621440);    // 2048x512  f16 (2 MB)
    _Float16* w1f_h    = (_Float16*)(ws + 4718592);    // 2048x1024 f16 (4 MB)
    _Float16* w1r_h    = (_Float16*)(ws + 8912896);    // 2048x1024 f16 (4 MB)
    _Float16* pooled_h = (_Float16*)(ws + 13107200);   // 2048x512  f16 (2 MB)
    _Float16* o0_h     = (_Float16*)(ws + 15204352);   // 2048x1024 f16 (4 MB)
    float*    o1_f     = (float*)   (ws + 19398656);   // 2048x1024 f32 (8 MB)

    auto cvt = [&](const float* s, _Float16* d, int n) {
        int n4 = n / 4;
        f32_to_f16_kernel<<<(n4 + 255) / 256, 256, 0, stream>>>(s, d, n4);
    };
    cvt(conv_w,   convw_h, 512 * 512);
    cvt(w_ih_l0f, w0f_h,   2048 * 512);
    cvt(w_ih_l0r, w0r_h,   2048 * 512);
    cvt(w_ih_l1f, w1f_h,   2048 * 1024);
    cvt(w_ih_l1r, w1r_h,   2048 * 1024);

    size_t smem1 = 66560 + 132096 + (64 + 64 + 64 + 256) * 4;  // 200,448 B (< 320 KB)
    conv_se_pool_kernel<<<2048, 256, smem1, stream>>>(
        x, convw_h, conv_b, se_w, se_b, pooled_h);

    lstm_step_kernel<<<1024, 256, 0, stream>>>(
        pooled_h, w0f_h, w0r_h,
        b_ih_l0f, b_hh_l0f, b_ih_l0r, b_hh_l0r,
        512, o0_h, nullptr, 0);

    lstm_step_kernel<<<1024, 256, 0, stream>>>(
        o0_h, w1f_h, w1r_h,
        b_ih_l1f, b_hh_l1f, b_ih_l1r, b_hh_l1r,
        1024, nullptr, o1_f, 1);

    cls_kernel<<<256, 256, 0, stream>>>(o1_f, cls_w, cls_b, out);
}
